// MultiHeadAttention_23029614641158
// MI455X (gfx1250) — matrix operand order repaired and checked by reading, not compiled, not run on hardware
//
#include <hip/hip_runtime.h>

// MHA forward for MI455X (gfx1250), wave32 + v_wmma_f32_16x16x32_f16.
// B=4, S=2048, D=512, H=8, depth=64.
// d_out = [ output(4*2048*512 f32) | attn_w(4*8*2048*2048 f32) ]
// d_ws  = Qh f16[B,H,S,64] | Kh f16[B,H,S,64] | Vt f16[B,H,64,S] | AOut f16[B,S,512]  (32 MB)

#define Bn    4
#define Sn    2048
#define Dn    512
#define Hn    8
#define DEPTH 64

typedef __attribute__((ext_vector_type(16))) _Float16 v16h;
typedef __attribute__((ext_vector_type(8)))  float    v8f;

__device__ __forceinline__ v8f wmma_f16(v16h a, v16h b, v8f c) {
  return __builtin_amdgcn_wmma_f32_16x16x32_f16(false, a, false, b, (short)0, c,
                                                false, false);
}

// ---- fragment loaders (per CDNA5 ISA 7.12.2 layouts) ----
// A 16x32 f16: lane<16: M=lane, K = {k0..k0+7, k0+16..k0+23}
//              lane>=16: M=lane-16, K = {k0+8..k0+15, k0+24..k0+31}
__device__ __forceinline__ v16h frag_a_f32(const float* __restrict__ A, int lda, int k0) {
  int lane = threadIdx.x & 31;
  int m = lane & 15, hf = lane >> 4;
  const float* p = A + (long)m * lda + k0 + hf * 8;
  v16h r;
#pragma unroll
  for (int i = 0; i < 8; ++i) r[i] = (_Float16)p[i];
#pragma unroll
  for (int i = 0; i < 8; ++i) r[8 + i] = (_Float16)p[16 + i];
  return r;
}

__device__ __forceinline__ v16h frag_a_f16(const _Float16* __restrict__ A, int lda, int k0) {
  int lane = threadIdx.x & 31;
  int m = lane & 15, hf = lane >> 4;
  const _Float16* p = A + (long)m * lda + k0 + hf * 8;
  v16h r;
#pragma unroll
  for (int i = 0; i < 8; ++i) r[i] = p[i];
#pragma unroll
  for (int i = 0; i < 8; ++i) r[8 + i] = p[16 + i];
  return r;
}

// A-fragment sourced from LDS f32 (row stride Sn)
__device__ __forceinline__ v16h frag_a_lds(const float* P, int k0) {
  int lane = threadIdx.x & 31;
  int m = lane & 15, hf = lane >> 4;
  const float* p = P + m * Sn + k0 + hf * 8;
  v16h r;
#pragma unroll
  for (int i = 0; i < 8; ++i) r[i] = (_Float16)p[i];
#pragma unroll
  for (int i = 0; i < 8; ++i) r[8 + i] = (_Float16)p[16 + i];
  return r;
}

// B 32x16: B[k][n] taken from row-major rows: B[k][n] = Rows[n][k]
// lane<16: col n=lane, K = {k0..k0+7, k0+16..k0+23} ; lane>=16: col n=lane-16, K = {k0+8..k0+15, k0+24..k0+31}
__device__ __forceinline__ v16h frag_b_f32(const float* __restrict__ Rows, int ldr, int k0) {
  int lane = threadIdx.x & 31;
  int n = lane & 15, hf = lane >> 4;
  const float* p = Rows + (long)n * ldr + k0 + hf * 8;
  v16h r;
#pragma unroll
  for (int i = 0; i < 8; ++i) r[i] = (_Float16)p[i];
#pragma unroll
  for (int i = 0; i < 8; ++i) r[8 + i] = (_Float16)p[16 + i];
  return r;
}

__device__ __forceinline__ v16h frag_b_f16(const _Float16* __restrict__ Rows, int ldr, int k0) {
  int lane = threadIdx.x & 31;
  int n = lane & 15, hf = lane >> 4;
  const _Float16* p = Rows + (long)n * ldr + k0 + hf * 8;
  v16h r;
#pragma unroll
  for (int i = 0; i < 8; ++i) r[i] = p[i];
#pragma unroll
  for (int i = 0; i < 8; ++i) r[8 + i] = p[16 + i];
  return r;
}

// ---- kernel 1: Q/K/V projections  C[r,n] = X[r,:] . W[n,:] + b[n] ----
// grid = (8, 64, 3), block = 256 (8 waves). Wave -> 16x64 strip.
// z=0 -> Qh [B,H,S,64], z=1 -> Kh [B,H,S,64], z=2 -> Vt [B,H,64,S] (transposed).
__global__ void __launch_bounds__(256)
mha_proj_kernel(const float* __restrict__ query, const float* __restrict__ key,
                const float* __restrict__ value,
                const float* __restrict__ Wq, const float* __restrict__ bq,
                const float* __restrict__ Wk, const float* __restrict__ bk,
                const float* __restrict__ Wv, const float* __restrict__ bv,
                _Float16* __restrict__ Qh, _Float16* __restrict__ Kh,
                _Float16* __restrict__ Vt) {
  int wave = threadIdx.x >> 5, lane = threadIdx.x & 31;
  int row0 = (blockIdx.y * 8 + wave) * 16;
  int n0   = blockIdx.x * 64;
  int z    = blockIdx.z;

  const float* X    = (z == 0) ? query : (z == 1) ? key : value;
  const float* W    = (z == 0) ? Wq    : (z == 1) ? Wk  : Wv;
  const float* bias = (z == 0) ? bq    : (z == 1) ? bk  : bv;
  _Float16*    dst  = (z == 0) ? Qh    : (z == 1) ? Kh  : Vt;

  v8f acc[4] = {};
  const float* Arow = X + (long)row0 * Dn;
  for (int k0 = 0; k0 < Dn; k0 += 32) {
    v16h a = frag_a_f32(Arow, Dn, k0);
#pragma unroll
    for (int j = 0; j < 4; ++j) {
      v16h b = frag_b_f32(W + (long)(n0 + j * 16) * Dn, Dn, k0);
      acc[j] = wmma_f16(a, b, acc[j]);
    }
  }

  int m_off = (lane >> 4) * 8;
  int n     = lane & 15;
  int h     = n0 >> 6;  // strip is 64-wide & 64-aligned -> single head
#pragma unroll
  for (int j = 0; j < 4; ++j) {
    int   d  = j * 16 + n;
    float bv_ = bias[n0 + j * 16 + n];
#pragma unroll
    for (int i = 0; i < 8; ++i) {
      int r = row0 + m_off + i;
      int b_ = r >> 11, s = r & (Sn - 1);
      _Float16 val = (_Float16)(acc[j][i] + bv_);
      if (z < 2)
        dst[((long)(b_ * Hn + h) * Sn + s) * DEPTH + d] = val;
      else
        dst[((long)(b_ * Hn + h) * DEPTH + d) * Sn + s] = val;
    }
  }
}

// ---- kernel 2: per (b,h,q-tile): scores -> softmax -> attn_w + P.V ----
// grid = B*H*(S/16) = 4096 blocks, block = 256 (8 waves).
__global__ void __launch_bounds__(256)
mha_attn_kernel(const _Float16* __restrict__ Qh, const _Float16* __restrict__ Kh,
                const _Float16* __restrict__ Vt, const float* __restrict__ mask,
                float* __restrict__ attn, _Float16* __restrict__ AOut) {
  __shared__ float P[16 * Sn];  // 128 KB score strip (reused as reduction buf)

  int wave = threadIdx.x >> 5, lane = threadIdx.x & 31;
  int bh = blockIdx.x >> 7;        // S/16 = 128 q-tiles per (b,h)
  int qt = blockIdx.x & 127;
  int b_ = bh >> 3, h = bh & 7;
  int q0 = qt * 16;

  const _Float16* Qbase = Qh + (long)bh * Sn * DEPTH;
  const _Float16* Kbase = Kh + (long)bh * Sn * DEPTH;
  const _Float16* Vbase = Vt + (long)bh * DEPTH * Sn;

  int m_off = (lane >> 4) * 8;
  int ncol  = lane & 15;

  // Q fragments (d = 0..31 and 32..63), reused for all 16 key tiles
  v16h aQ0 = frag_a_f16(Qbase + (long)q0 * DEPTH, DEPTH, 0);
  v16h aQ1 = frag_a_f16(Qbase + (long)q0 * DEPTH, DEPTH, 32);

  // ---- phase 1: scores = (Q.K^T)*scale + mask, into LDS ----
  for (int nt = 0; nt < 16; ++nt) {
    int kc0 = wave * 256 + nt * 16;
    v8f acc = {};
    v16h b0 = frag_b_f16(Kbase + (long)kc0 * DEPTH, DEPTH, 0);
    acc = wmma_f16(aQ0, b0, acc);
    v16h b1 = frag_b_f16(Kbase + (long)kc0 * DEPTH, DEPTH, 32);
    acc = wmma_f16(aQ1, b1, acc);
#pragma unroll
    for (int i = 0; i < 8; ++i) {
      int mrow = m_off + i;
      int q = q0 + mrow, kk = kc0 + ncol;
      P[mrow * Sn + kk] =
          acc[i] * 0.125f + mask[((long)b_ * Sn + q) * Sn + kk];
    }
  }
  __syncthreads();

  // ---- phase 2: softmax per row (wave handles 2 rows), write attn_w ----
  for (int rr = 0; rr < 2; ++rr) {
    int row = wave * 2 + rr;
    float mx = -1e30f;
    for (int c = lane; c < Sn; c += 32) mx = fmaxf(mx, P[row * Sn + c]);
#pragma unroll
    for (int o = 16; o > 0; o >>= 1) mx = fmaxf(mx, __shfl_xor(mx, o, 32));
    float sum = 0.f;
    for (int c = lane; c < Sn; c += 32) {
      float e = __expf(P[row * Sn + c] - mx);
      P[row * Sn + c] = e;
      sum += e;
    }
#pragma unroll
    for (int o = 16; o > 0; o >>= 1) sum += __shfl_xor(sum, o, 32);
    float inv = 1.0f / sum;
    long abase = ((long)bh * Sn + (q0 + row)) * Sn;
    for (int c = lane; c < Sn; c += 32) {
      float p = P[row * Sn + c] * inv;
      P[row * Sn + c] = p;
      attn[abase + c] = p;
    }
  }
  __syncthreads();

  // ---- phase 3: attn_out tile = P(16xS) . V(Sx64), k split across waves ----
  v8f oacc[4] = {};
  for (int t = 0; t < 8; ++t) {
    int k0 = wave * 256 + t * 32;
    v16h aP = frag_a_lds(P, k0);
#pragma unroll
    for (int j = 0; j < 4; ++j) {
      v16h bV = frag_b_f16(Vbase + (long)(j * 16) * Sn, Sn, k0);
      oacc[j] = wmma_f16(aP, bV, oacc[j]);
    }
  }
  __syncthreads();  // all P reads done; reuse P as reduction buffer
#pragma unroll
  for (int j = 0; j < 4; ++j)
#pragma unroll
    for (int i = 0; i < 8; ++i)
      P[(wave * 4 + j) * 256 + lane * 8 + i] = oacc[j][i];
  __syncthreads();

  if (wave < 4) {
    int j = wave;  // depth tile
    v8f s = {};
#pragma unroll
    for (int w = 0; w < 8; ++w)
#pragma unroll
      for (int i = 0; i < 8; ++i)
        s[i] += P[(w * 4 + j) * 256 + lane * 8 + i];
#pragma unroll
    for (int i = 0; i < 8; ++i) {
      int q = q0 + m_off + i;
      int col = h * DEPTH + j * 16 + ncol;  // [B,S,H*depth] layout
      AOut[((long)b_ * Sn + q) * Dn + col] = (_Float16)s[i];
    }
  }
}

// ---- kernel 3: output = AOut . Wo^T + bo  (f32 out) ----
// grid = (8, 64), block = 256.
__global__ void __launch_bounds__(256)
mha_out_kernel(const _Float16* __restrict__ AOut, const float* __restrict__ Wo,
               const float* __restrict__ bo, float* __restrict__ out) {
  int wave = threadIdx.x >> 5, lane = threadIdx.x & 31;
  int row0 = (blockIdx.y * 8 + wave) * 16;
  int n0   = blockIdx.x * 64;

  v8f acc[4] = {};
  const _Float16* Arow = AOut + (long)row0 * Dn;
  for (int k0 = 0; k0 < Dn; k0 += 32) {
    v16h a = frag_a_f16(Arow, Dn, k0);
#pragma unroll
    for (int j = 0; j < 4; ++j) {
      v16h b = frag_b_f32(Wo + (long)(n0 + j * 16) * Dn, Dn, k0);
      acc[j] = wmma_f16(a, b, acc[j]);
    }
  }
  int m_off = (lane >> 4) * 8, n = lane & 15;
#pragma unroll
  for (int j = 0; j < 4; ++j) {
    float bv_ = bo[n0 + j * 16 + n];
#pragma unroll
    for (int i = 0; i < 8; ++i) {
      int r = row0 + m_off + i;
      out[(long)r * Dn + n0 + j * 16 + n] = acc[j][i] + bv_;
    }
  }
}

extern "C" void kernel_launch(void* const* d_in, const int* in_sizes, int n_in,
                              void* d_out, int out_size, void* d_ws, size_t ws_size,
                              hipStream_t stream) {
  (void)in_sizes; (void)n_in; (void)out_size; (void)ws_size;
  const float* query = (const float*)d_in[0];
  const float* key   = (const float*)d_in[1];
  const float* value = (const float*)d_in[2];
  const float* mask  = (const float*)d_in[3];
  const float* Wq_w  = (const float*)d_in[4];
  const float* Wq_b  = (const float*)d_in[5];
  const float* Wk_w  = (const float*)d_in[6];
  const float* Wk_b  = (const float*)d_in[7];
  const float* Wv_w  = (const float*)d_in[8];
  const float* Wv_b  = (const float*)d_in[9];
  const float* Wo_w  = (const float*)d_in[10];
  const float* Wo_b  = (const float*)d_in[11];

  float* out  = (float*)d_out;
  float* attn = out + (long)Bn * Sn * Dn;

  const long perQK = (long)Bn * Hn * Sn * DEPTH;  // 4,194,304 f16
  _Float16* Qh   = (_Float16*)d_ws;
  _Float16* Kh   = Qh + perQK;
  _Float16* Vt   = Kh + perQK;
  _Float16* AOut = Vt + perQK;

  dim3 g1(Dn / 64, (Bn * Sn) / (8 * 16), 3);
  mha_proj_kernel<<<g1, 256, 0, stream>>>(query, key, value, Wq_w, Wq_b, Wk_w,
                                          Wk_b, Wv_w, Wv_b, Qh, Kh, Vt);

  dim3 g2(Bn * Hn * (Sn / 16));
  mha_attn_kernel<<<g2, 256, 0, stream>>>(Qh, Kh, Vt, mask, attn, AOut);

  dim3 g3(Dn / 64, (Bn * Sn) / (8 * 16));
  mha_out_kernel<<<g3, 256, 0, stream>>>(AOut, Wo_w, Wo_b, out);
}
